// SelfAttention_68650757259942
// MI455X (gfx1250) — matrix operand order repaired and checked by reading, not compiled, not run on hardware
//
#include <hip/hip_runtime.h>

// ---------------------------------------------------------------------------
// Self-attention block (QKV proj + RoPE + causal flash attention + out proj)
// for MI455X / gfx1250: all GEMM-shaped math on v_wmma_f32_16x16x32_f16.
// Shapes fixed by the reference: B=2, T=2048, D=1024, H=16, HD=64.
// ---------------------------------------------------------------------------

typedef _Float16 f16;
typedef __attribute__((ext_vector_type(16))) _Float16 v16h;
typedef __attribute__((ext_vector_type(8)))  float    v8f;

static constexpr int kB  = 2;
static constexpr int kT  = 2048;
static constexpr int kD  = 1024;
static constexpr int kH  = 16;
static constexpr int kHD = 64;

#define WMMA_F16(a, b, c) \
  __builtin_amdgcn_wmma_f32_16x16x32_f16(false, (a), false, (b), (short)0, (c), false, false)

// Build a 16-half WMMA fragment from two 16-byte chunks (all call sites are
// 16B aligned: every half-offset used is a multiple of 8).
__device__ __forceinline__ v16h load_frag2(const f16* p0, const f16* p1) {
  v16h r;
  uint4* u = reinterpret_cast<uint4*>(&r);
  u[0] = *reinterpret_cast<const uint4*>(p0);
  u[1] = *reinterpret_cast<const uint4*>(p1);
  return r;
}

__device__ __forceinline__ float redmax16(float v) {  // replicate over 16-lane half
  v = fmaxf(v, __shfl_xor(v, 1, 32));
  v = fmaxf(v, __shfl_xor(v, 2, 32));
  v = fmaxf(v, __shfl_xor(v, 4, 32));
  v = fmaxf(v, __shfl_xor(v, 8, 32));
  return v;
}
__device__ __forceinline__ float redsum16(float v) {
  v += __shfl_xor(v, 1, 32);
  v += __shfl_xor(v, 2, 32);
  v += __shfl_xor(v, 4, 32);
  v += __shfl_xor(v, 8, 32);
  return v;
}

// ---------------------------------------------------------------------------
// GEMM: C[M,N] = A[M,K] (f32) * Bw[K,N] (f32) + bias[N], f16 WMMA compute,
// f32 accumulate. Block tile 64x128, K-step 32, 8 waves (256 thr, wave32).
// Each wave: one 16-row A-frag reused by 4 WMMAs (16x64 output strip).
// Global loads for tile k+1 are software-pipelined in registers across the
// compute phase of tile k. Requires M%64==0, N%128==0, K%32==0.
// ---------------------------------------------------------------------------
__global__ __launch_bounds__(256, 2)
void gemm_bias_f32_wmma(const float* __restrict__ A, const float* __restrict__ Bw,
                        const float* __restrict__ bias, float* __restrict__ C,
                        int M, int N, int K) {
  constexpr int BM = 64, BN = 128, BK = 32;
  constexpr int ASS = 40;  // padded LDS strides (halves) to spread banks
  constexpr int BSS = 40;
  __shared__ alignas(16) f16 As[BM * ASS];   // As[m][k]
  __shared__ alignas(16) f16 Bs[BN * BSS];   // Bs[n][k]  (transposed stage)

  const int tid  = threadIdx.x;
  const int m0   = blockIdx.y * BM;
  const int n0   = blockIdx.x * BN;
  const int wave = tid >> 5, lane = tid & 31;
  const int hi   = lane >> 4, l16 = lane & 15;
  const int mT   = wave & 3;          // wave's 16-row tile (0..3)
  const int nT0  = (wave >> 2) * 4;   // wave's first 16-col tile (0 or 4)

  v8f acc[4] = {{}, {}, {}, {}};

  const int arow = tid >> 2, acb = (tid & 3) * 8;   // A stage: 64 rows x 32 k
  const int brow = tid >> 3, bnb = (tid & 7) * 16;  // B stage: 32 k x 128 n

  const float* aptr = &A[(size_t)(m0 + arow) * K + acb];
  const float* bptr = &Bw[(size_t)brow * N + n0 + bnb];

  // ---- software pipeline: preload tile 0 into registers ----
  float4 ar0, ar1, br0, br1, br2, br3;
  {
    const float4* ap = reinterpret_cast<const float4*>(aptr);
    ar0 = ap[0]; ar1 = ap[1];
    const float4* bp = reinterpret_cast<const float4*>(bptr);
    br0 = bp[0]; br1 = bp[1]; br2 = bp[2]; br3 = bp[3];
  }

  for (int k0 = 0; k0 < K; k0 += BK) {
    // ---- stage current tile registers -> LDS (fp32 -> f16) ----
    {
      f16* d = &As[arow * ASS + acb];
      d[0] = (f16)ar0.x; d[1] = (f16)ar0.y; d[2] = (f16)ar0.z; d[3] = (f16)ar0.w;
      d[4] = (f16)ar1.x; d[5] = (f16)ar1.y; d[6] = (f16)ar1.z; d[7] = (f16)ar1.w;
      const float bv[16] = {br0.x, br0.y, br0.z, br0.w, br1.x, br1.y, br1.z, br1.w,
                            br2.x, br2.y, br2.z, br2.w, br3.x, br3.y, br3.z, br3.w};
#pragma unroll
      for (int j = 0; j < 16; ++j) Bs[(bnb + j) * BSS + brow] = (f16)bv[j];
    }
    __syncthreads();

    // ---- issue next tile's global loads (overlap with WMMA below) ----
    if (k0 + BK < K) {
      const float4* ap = reinterpret_cast<const float4*>(aptr + k0 + BK);
      ar0 = ap[0]; ar1 = ap[1];
      const float4* bp = reinterpret_cast<const float4*>(bptr + (size_t)(k0 + BK) * N);
      br0 = bp[0]; br1 = bp[1]; br2 = bp[2]; br3 = bp[3];
      if (k0 + 2 * BK < K) {  // L2 prefetch one tile further (global_prefetch_b8)
        __builtin_prefetch(aptr + k0 + 2 * BK, 0, 1);
        __builtin_prefetch(bptr + (size_t)(k0 + 2 * BK) * N, 0, 1);
      }
    }

    // ---- compute: 1 A-frag, 4 B-frags, 4 WMMAs ----
    // A frag: lane row = mT*16+l16; halves k = i + 16*hi (contig)
    const f16* arp = &As[(mT * 16 + l16) * ASS];
    const v16h af = load_frag2(arp + 16 * hi, arp + 16 * hi + 8);
#pragma unroll
    for (int nti = 0; nti < 4; ++nti) {
      // B frag: lane col n = (nT0+nti)*16+l16; halves k = i + 16*hi (contig)
      const f16* bp = &Bs[((nT0 + nti) * 16 + l16) * BSS + 16 * hi];
      const v16h bf = load_frag2(bp, bp + 8);
      acc[nti] = WMMA_F16(af, bf, acc[nti]);
    }
    __syncthreads();
  }

#pragma unroll
  for (int nti = 0; nti < 4; ++nti) {  // C layout: rows r+8*hi, col = lane&15
    const int col = n0 + (nT0 + nti) * 16 + l16;
    const float bv = bias[col];
#pragma unroll
    for (int r = 0; r < 8; ++r) {
      const int row = m0 + mT * 16 + r + 8 * hi;
      C[(size_t)row * N + col] = acc[nti][r] + bv;
    }
  }
}

// ---------------------------------------------------------------------------
// RoPE + layout pack: qkv f32 [B*T, 3D] ->
//   qh,kh f16 [B,H,T,HD] (RoPE applied), vT f16 [B,H,HD,T] (transposed).
// One thread per (b,h,t,d) with d in [0,32): handles the RoPE pair (d, d+32).
// ---------------------------------------------------------------------------
__global__ __launch_bounds__(256)
void rope_pack(const float* __restrict__ qkv, f16* __restrict__ qh,
               f16* __restrict__ kh, f16* __restrict__ vT) {
  const int idx = blockIdx.x * 256 + threadIdx.x;   // 2^21 threads total
  const int d = idx & 31;
  const int h = (idx >> 5) & (kH - 1);
  const int t = (idx >> 9) & (kT - 1);
  const int b = idx >> 20;

  const size_t base = (size_t)(b * kT + t) * (3 * kD) + h * kHD;
  const float q1 = qkv[base + d],            q2 = qkv[base + d + 32];
  const float k1 = qkv[base + kD + d],       k2 = qkv[base + kD + d + 32];
  const float v1 = qkv[base + 2 * kD + d],   v2 = qkv[base + 2 * kD + d + 32];

  const float rate = __powf(10000.0f, -(float)d * (1.0f / 32.0f));
  float sn, cs;
  __sincosf((float)t * rate, &sn, &cs);

  const size_t qko = ((size_t)(b * kH + h) * kT + t) * kHD + d;
  qh[qko]      = (f16)(q1 * cs - q2 * sn);
  qh[qko + 32] = (f16)(q1 * sn + q2 * cs);
  kh[qko]      = (f16)(k1 * cs - k2 * sn);
  kh[qko + 32] = (f16)(k1 * sn + k2 * cs);

  const size_t vto = ((size_t)(b * kH + h) * kHD + d) * kT + t;
  vT[vto]           = (f16)v1;
  vT[vto + 32 * kT] = (f16)v2;
}

// ---------------------------------------------------------------------------
// Causal flash attention. One wave per 16-query tile; 4 waves/block.
// Per 32-key block: 4 WMMAs for S = Q K^T (two 16x16 tiles, K=64 in two
// k-steps), online softmax (row stats via shfl_xor over 16-lane halves),
// P bounced through LDS (C-layout -> A-layout), 4 WMMAs for O += P V.
// Writes y f32 [B,T,D] (already /l), ready for the output projection GEMM.
// ---------------------------------------------------------------------------
__global__ __launch_bounds__(128)
void flash_attn_wmma(const f16* __restrict__ qh, const f16* __restrict__ kh,
                     const f16* __restrict__ vT, float* __restrict__ y) {
  __shared__ alignas(16) f16 Ps[4][16 * 32];  // per-wave P tile (16q x 32k)

  const int lane = threadIdx.x & 31, wave = threadIdx.x >> 5;
  const int hi = lane >> 4, l16 = lane & 15;
  const int gtile = blockIdx.x * 4 + wave;       // 4096 q-tiles total
  const int qt = gtile & (kT / 16 - 1);          // q-tile within (b,h)
  const int bh = gtile >> 7;                     // T/16 == 128
  const int b = bh >> 4, h = bh & (kH - 1);

  const f16* qb = qh + ((size_t)bh * kT + qt * 16) * kHD;
  const f16* kb = kh + (size_t)bh * kT * kHD;
  const f16* vb = vT + (size_t)bh * kHD * kT;

  // Q fragments (two k-steps of 32 dims), pre-scaled by 1/sqrt(HD)
  v16h q[2];
#pragma unroll
  for (int s = 0; s < 2; ++s) {
    const f16* qp = qb + l16 * kHD + 32 * s;
    q[s] = load_frag2(qp + 16 * hi, qp + 16 * hi + 8);
#pragma unroll
    for (int i = 0; i < 16; ++i) q[s][i] *= (f16)0.125f;
  }

  v8f o0 = {}, o1 = {}, o2 = {}, o3 = {};
  float mi[8], li[8];
#pragma unroll
  for (int r = 0; r < 8; ++r) { mi[r] = -1e30f; li[r] = 0.0f; }

  const int nblk = (qt * 16 + 15) / 32 + 1;
  for (int j = 0; j < nblk; ++j) {
    const int k0 = j * 32;

    // S tiles: queries = rows, keys = lanes
    v8f s01[2];
#pragma unroll
    for (int c = 0; c < 2; ++c) {
      const f16* kp = kb + (size_t)(k0 + 16 * c + l16) * kHD;
      v16h bk0 = load_frag2(kp + 16 * hi, kp + 16 * hi + 8);
      v16h bk1 = load_frag2(kp + 32 + 16 * hi, kp + 40 + 16 * hi);
      v8f s = {};
      s = WMMA_F16(q[0], bk0, s);
      s = WMMA_F16(q[1], bk1, s);
      s01[c] = s;
    }

    if (j == nblk - 1) {  // causal mask on the diagonal block
#pragma unroll
      for (int r = 0; r < 8; ++r) {
        const int row = qt * 16 + r + 8 * hi;
        if (k0 + l16 > row)      s01[0][r] = -1e30f;
        if (k0 + 16 + l16 > row) s01[1][r] = -1e30f;
      }
    }

    // online softmax; stage P into LDS in row-major [16 x 32] f16
#pragma unroll
    for (int r = 0; r < 8; ++r) {
      float sm = redmax16(fmaxf(s01[0][r], s01[1][r]));
      const float mnew = fmaxf(mi[r], sm);
      const float co = __expf(mi[r] - mnew);
      const float p0 = __expf(s01[0][r] - mnew);
      const float p1 = __expf(s01[1][r] - mnew);
      li[r] = li[r] * co + redsum16(p0 + p1);
      mi[r] = mnew;
      o0[r] *= co; o1[r] *= co; o2[r] *= co; o3[r] *= co;
      f16* pr = &Ps[wave][(r + 8 * hi) * 32 + l16];
      pr[0]  = (f16)p0;
      pr[16] = (f16)p1;
    }
    // intra-wave LDS RAW: DS ops are in-order per wave; compiler emits waits.

    // P as A-fragment (16x32), V^T columns as B-fragments (contiguous keys)
    const f16* pp = &Ps[wave][l16 * 32];
    v16h pf = load_frag2(pp + 16 * hi, pp + 16 * hi + 8);
#pragma unroll
    for (int nt = 0; nt < 4; ++nt) {
      const f16* vp = vb + (size_t)(nt * 16 + l16) * kT + k0 + 16 * hi;
      v16h vf = load_frag2(vp, vp + 8);
      v8f* op = (nt == 0) ? &o0 : (nt == 1) ? &o1 : (nt == 2) ? &o2 : &o3;
      *op = WMMA_F16(pf, vf, *op);
    }
  }

#pragma unroll
  for (int r = 0; r < 8; ++r) {
    const float inv = 1.0f / li[r];
    const int t = qt * 16 + r + 8 * hi;
    float* yp = y + (size_t)(b * kT + t) * kD + h * kHD + l16;
    yp[0]  = o0[r] * inv;
    yp[16] = o1[r] * inv;
    yp[32] = o2[r] * inv;
    yp[48] = o3[r] * inv;
  }
}

// ---------------------------------------------------------------------------
// Launch: x@Wqkv -> rope/pack -> flash attention -> y@Wproj.
// Workspace layout (75.5 MB): [qkv f32 48MB | qh f16 8MB | kh 8MB | vT 8MB],
// with y (f32, 16MB) aliasing the qkv region (qkv is dead after rope_pack).
// ---------------------------------------------------------------------------
extern "C" void kernel_launch(void* const* d_in, const int* in_sizes, int n_in,
                              void* d_out, int out_size, void* d_ws, size_t ws_size,
                              hipStream_t stream) {
  (void)in_sizes; (void)n_in; (void)out_size; (void)ws_size;

  const float* x     = (const float*)d_in[0];
  const float* Wqkv  = (const float*)d_in[1];
  const float* bqkv  = (const float*)d_in[2];
  const float* Wproj = (const float*)d_in[3];
  const float* bproj = (const float*)d_in[4];
  float* out = (float*)d_out;

  const size_t SQKV = (size_t)kB * kT * 3 * kD * sizeof(float);      // 48 MB
  const size_t QN   = (size_t)kB * kH * kT * kHD;                    // f16 elems

  float* qkv = (float*)d_ws;
  f16* qh = (f16*)((char*)d_ws + SQKV);
  f16* kh = qh + QN;
  f16* vT = kh + QN;
  float* y = qkv;  // alias: qkv no longer needed once attention starts

  const int M = kB * kT;  // 4096

  // 1) qkv = x @ Wqkv + bqkv   (4096 x 3072 x 1024)
  gemm_bias_f32_wmma<<<dim3(3 * kD / 128, M / 64), 256, 0, stream>>>(
      x, Wqkv, bqkv, qkv, M, 3 * kD, kD);

  // 2) RoPE + f16 pack (+ V transpose)
  rope_pack<<<(kB * kT * kH * 32) / 256, 256, 0, stream>>>(qkv, qh, kh, vT);

  // 3) causal flash attention -> y [B,T,D] f32
  flash_attn_wmma<<<(kB * kH * (kT / 16)) / 4, 128, 0, stream>>>(qh, kh, vT, y);

  // 4) out = y @ Wproj + bproj   (4096 x 1024 x 1024)
  gemm_bias_f32_wmma<<<dim3(kD / 128, M / 64), 256, 0, stream>>>(
      y, Wproj, bproj, out, M, kD, kD);
}
